// AttentionBasedLSTM_53901839565264
// MI455X (gfx1250) — matrix operand order repaired and checked by reading, not compiled, not run on hardware
//
#include <hip/hip_runtime.h>
#include <hip/hip_bf16.h>
#include <math.h>

// ---- problem constants ----
#define BB 16
#define CC 512
#define HH 16
#define WW 64
#define TT 128
#define VV 600
#define EE 80
#define EHH 256      // encoder hidden
#define DHH 512      // decoder hidden
#define BH  (BB*HH)          // 256 encoder batch rows
#define LL  (BH*WW)          // 16384 encoder sequence rows flattened
#define PP  (HH*WW)          // 1024 attention positions per batch

typedef __bf16 bf16_t;
typedef __attribute__((ext_vector_type(16))) __bf16 v16bf;
typedef __attribute__((ext_vector_type(8)))  float  v8f;

#define DEV __device__ __forceinline__

DEV bf16_t f2b(float x) { return (bf16_t)x; }
DEV float  sigf(float x){ return 1.0f/(1.0f + expf(-x)); }

// =====================================================================
// Fragment-major packed layout: matrix tiled into 16x16(x32-K) WMMA tiles.
// Tile (outer, kt) stored contiguously: 32 lanes x 16 bf16 = 1KB, kt innermost.
//   B tile (KxN):  lane = half*16 + n_in_tile ; element e -> K = kt*32 + (e>>3)*16 + half*8 + (e&7)
//   A tile (MxK):  lane = half*16 + m ; element e -> K = kt*32 + (e>>3)*16 + half*8 + (e&7)
// Fragment load = one 32B contiguous vector load per lane (coalesced 1KB/wave).
// =====================================================================
DEV v16bf frag_packed(const bf16_t* P, int Kt, int outer, int kt, int lane) {
  return *(const v16bf*)(P + ((((size_t)outer*Kt + kt)*32 + lane) << 4));
}
// dynamic A operand: f32 row-major 16xK, converted on load
DEV v16bf frag_a_f32(const float* A, int lda, int m0, int k0, int lane) {
  int half = lane >> 4;
  const float* row = A + (size_t)(m0 + (lane & 15)) * lda + k0;
  v16bf a;
#pragma unroll
  for (int e = 0; e < 16; ++e) { int g = e >> 3, o = e & 7; a[e] = f2b(row[g*16 + half*8 + o]); }
  return a;
}
// dynamic A operand split across two f32 sources (concat along K)
DEV v16bf frag_a_concat(const float* A1, int K1, const float* A2, int K2, int k0, int lane) {
  int half = lane >> 4, m = lane & 15;
  v16bf a;
#pragma unroll
  for (int e = 0; e < 16; ++e) {
    int g = e >> 3, o = e & 7;
    int k = k0 + g*16 + half*8 + o;
    float v = (k < K1) ? A1[(size_t)m*K1 + k] : A2[(size_t)m*K2 + (k - K1)];
    a[e] = f2b(v);
  }
  return a;
}
DEV v8f wmma_bf16(v16bf a, v16bf b, v8f c) {
  return __builtin_amdgcn_wmma_f32_16x16x32_bf16(false, a, false, b, (short)0, c, false, false);
}

// ---------- weight -> packed-B fragment layout; src is (N rows x Ksrc cols) = W[n][k] ----------
__global__ __launch_bounds__(256) void k_wpack(const float* src, bf16_t* dst, int N, int Ksrc, int Kpad) {
  int idx = blockIdx.x*256 + threadIdx.x;
  int numKt = Kpad >> 5;
  if (idx >= Kpad*N) return;
  int tile = idx >> 9, r = idx & 511;
  int lane = r >> 4, e = r & 15;
  int nt = tile / numKt, kt = tile % numKt;
  int half = lane >> 4;
  int n = nt*16 + (lane & 15);
  int k = kt*32 + (e >> 3)*16 + half*8 + (e & 7);
  dst[idx] = (k < Ksrc) ? f2b(src[(size_t)n*Ksrc + k]) : (bf16_t)0.0f;
}

// ---------- pack features -> packed-A fragment layout, logical x[m = w*256+bh][k = c] ----------
__global__ __launch_bounds__(256) void k_pack_x(const float* feat, bf16_t* xp) {
  int idx = blockIdx.x*256 + threadIdx.x;
  if (idx >= LL*CC) return;
  const int numKt = CC/32;                 // 16
  int tile = idx >> 9, r = idx & 511;
  int lane = r >> 4, e = r & 15;
  int mt = tile / numKt, kt = tile % numKt;
  int half = lane >> 4;
  int m = mt*16 + (lane & 15);
  int k = kt*32 + (e >> 3)*16 + half*8 + (e & 7);
  int w = m >> 8, bh = m & 255, b = bh >> 4, h = bh & 15;
  xp[idx] = f2b(feat[(((size_t)b*CC + k)*HH + h)*WW + w]);
}

// ---------- generic packed-operand WMMA GEMM: C(MxN) = A*B + bias1 + bias2 ; wave = 16x64 ----------
__global__ __launch_bounds__(256) void k_gemm_pack(
  const bf16_t* Ap, const bf16_t* Bp,
  const float* bias1, const float* bias2,
  float* C, int ldc, int Mtiles, int Ngroups, int Kt)
{
  int wid = (blockIdx.x * blockDim.x + threadIdx.x) >> 5;
  int lane = threadIdx.x & 31;
  if (wid >= Mtiles * Ngroups) return;
  int mt  = wid / Ngroups;
  int nt0 = (wid % Ngroups) * 4;
  v8f a0 = {}, a1 = {}, a2 = {}, a3 = {};
  for (int kt = 0; kt < Kt; ++kt) {
    v16bf af = frag_packed(Ap, Kt, mt, kt, lane);
    a0 = wmma_bf16(af, frag_packed(Bp, Kt, nt0 + 0, kt, lane), a0);
    a1 = wmma_bf16(af, frag_packed(Bp, Kt, nt0 + 1, kt, lane), a1);
    a2 = wmma_bf16(af, frag_packed(Bp, Kt, nt0 + 2, kt, lane), a2);
    a3 = wmma_bf16(af, frag_packed(Bp, Kt, nt0 + 3, kt, lane), a3);
  }
  int half = lane >> 4, nn = lane & 15;
#pragma unroll
  for (int j = 0; j < 4; ++j) {
    int n = (nt0 + j)*16 + nn;
    float bv = (bias1 ? bias1[n] : 0.0f) + (bias2 ? bias2[n] : 0.0f);
    v8f acc = (j == 0) ? a0 : (j == 1) ? a1 : (j == 2) ? a2 : a3;
#pragma unroll
    for (int r = 0; r < 8; ++r)
      C[(size_t)(mt*16 + half*8 + r)*ldc + n] = acc[r] + bv;
  }
}

// ---------- init encoder h,c with Vh0/Vc0 broadcast ----------
__global__ __launch_bounds__(256) void k_init_hc(const float* Vh0, const float* Vc0, float* h, float* c) {
  int idx = blockIdx.x*256 + threadIdx.x;
  if (idx >= 2*BH*EHH) return;
  int dir = idx >> 16, j = idx & 255;
  h[idx] = Vh0[dir*EHH + j];
  c[idx] = Vc0[dir*EHH + j];
}

// ---------- one encoder scan step (both directions); fused WMMA gates + LSTM cell ----------
// grid = 128: bid = dir(1b) | mt(4b) | g(2b).  Block owns rows mt*16..+15, gate cols j in [g*64,g*64+64)
__global__ __launch_bounds__(256) void k_enc_step(
  const float* Gf, const float* Gb, const bf16_t* WhhfP, const bf16_t* WhhbP,
  const float* hprev, float* hnext, float* ccur, float* hs, int s)
{
  int bid = blockIdx.x;
  int dir = bid & 1;
  int mt  = (bid >> 1) & 15;
  int g   = bid >> 5;
  int t_eff = dir ? (63 - s) : s;
  const float* G = (dir ? Gb : Gf) + (size_t)t_eff * BH * 1024;
  const bf16_t* Wt = dir ? WhhbP : WhhfP;   // packed, Kt = 8
  const float* hp = hprev + (size_t)dir*BH*EHH;
  float* hn = hnext + (size_t)dir*BH*EHH;
  float* cd = ccur  + (size_t)dir*BH*EHH;

  __shared__ float ldsg[16*256];   // 16 rows x (4 gate chunks x 64)
  int tid = threadIdx.x, lane = tid & 31, wave = tid >> 5;

#pragma unroll
  for (int ti = 0; ti < 2; ++ti) {
    int idx = wave*2 + ti;                 // 0..15 tiles
    int q = idx >> 2, u = idx & 3;         // q = gate chunk (i/f/g/o)
    int nt = (q*256 + g*64 + u*16) >> 4;   // global n-tile
    v8f acc = {};
    for (int kt = 0; kt < EHH/32; ++kt) {
      v16bf af = frag_a_f32(hp, EHH, mt*16, kt*32, lane);
      acc = wmma_bf16(af, frag_packed(Wt, EHH/32, nt, kt, lane), acc);
    }
    int half = lane >> 4, nn = lane & 15;
    int cl = q*64 + u*16 + nn;
#pragma unroll
    for (int r = 0; r < 8; ++r) {
      int row = half*8 + r;
      ldsg[row*256 + cl] = acc[r] + G[(size_t)(mt*16 + row)*1024 + (nt*16 + nn)];
    }
  }
  __syncthreads();
#pragma unroll
  for (int c4 = 0; c4 < 4; ++c4) {
    int cidx = c4*256 + tid;               // 16 rows x 64 j = 1024 cells
    int row = cidx >> 6, jl = cidx & 63;
    int j = g*64 + jl;
    float iv = ldsg[row*256 +       jl];
    float fv = ldsg[row*256 +  64 + jl];
    float gv = ldsg[row*256 + 128 + jl];
    float ov = ldsg[row*256 + 192 + jl];
    size_t off = (size_t)(mt*16 + row)*EHH + j;
    float cp = cd[off];
    float cnew = sigf(fv)*cp + sigf(iv)*tanhf(gv);
    float hnew = sigf(ov)*tanhf(cnew);
    cd[off] = cnew;
    hn[off] = hnew;
    hs[(((size_t)dir*WW + t_eff)*BH + (mt*16 + row))*EHH + j] = hnew;
  }
}

// ---------- pack hs -> enc_out f32 (normal) + packed-A bf16 (for Wv GEMM) ----------
__global__ __launch_bounds__(256) void k_pack_enc(const float* hs, float* encf, bf16_t* encp) {
  int idx = blockIdx.x*256 + threadIdx.x;
  if (idx >= LL*512) return;
  {   // normal-layout element: row = b*1024 + h*64 + w, col d
    int rowE = idx >> 9, d = idx & 511;
    int b = rowE >> 10, p = rowE & 1023;
    int h = p >> 6, w = p & 63;
    int bh = b*16 + h;
    int dir = d >> 8, dd = d & 255;
    encf[idx] = hs[(((size_t)dir*WW + w)*BH + bh)*EHH + dd];
  }
  {   // packed-A-layout element (M=16384, K=512, Kt=16)
    const int numKt = 16;
    int tile = idx >> 9, r = idx & 511;
    int lane = r >> 4, e = r & 15;
    int mt = tile / numKt, kt = tile % numKt;
    int half = lane >> 4;
    int m = mt*16 + (lane & 15);
    int k = kt*32 + (e >> 3)*16 + half*8 + (e & 7);
    int b = m >> 10, p = m & 1023;
    int h = p >> 6, w = p & 63;
    int bh = b*16 + h;
    int dir = k >> 8, dd = k & 255;
    encp[idx] = f2b(hs[(((size_t)dir*WW + w)*BH + bh)*EHH + dd]);
  }
}

// ---------- decoder initial h0/c0 (mean over h of final states, [fwd|bwd]) ----------
__global__ __launch_bounds__(256) void k_h0c0(const float* hT, const float* cT, float* hdec, float* cdec) {
  int idx = blockIdx.x*256 + threadIdx.x;
  if (idx >= BB*DHH) return;
  int b = idx >> 9, d = idx & 511;
  int dir = d >> 8, dd = d & 255;
  float sh = 0.0f, sc = 0.0f;
  for (int h = 0; h < 16; ++h) {
    size_t off = ((size_t)dir*BH + b*16 + h)*EHH + dd;
    sh += hT[off]; sc += cT[off];
  }
  hdec[idx] = sh * (1.0f/16.0f);
  cdec[idx] = sc * (1.0f/16.0f);
}

// ---------- ctx0 = mean over 1024 positions ----------
__global__ __launch_bounds__(256) void k_ctx0(const float* encf, float* ctx) {
  int b = blockIdx.x, tid = threadIdx.x;
  for (int d = tid; d < 512; d += 256) {
    float s = 0.0f;
    const float* ef = encf + (size_t)b*PP*512 + d;
    for (int p = 0; p < PP; ++p) s += ef[(size_t)p*512];
    ctx[b*512 + d] = s * (1.0f/1024.0f);
  }
}

// ---------- small M=16 WMMA GEMM: C(16xN) = concat(A1,A2) @ Bp + bias ; act: 0=none 1=tanh ----------
__global__ __launch_bounds__(256) void k_gemm_m16(
  const float* A1, int K1, const float* A2, int K2,
  const bf16_t* Bp, int N, const float* bias, float* C, int act)
{
  int wid = (blockIdx.x*blockDim.x + threadIdx.x) >> 5;
  int lane = threadIdx.x & 31;
  if (wid >= (N >> 4)) return;
  int Kt = (K1 + K2) >> 5;
  v8f acc = {};
  for (int kt = 0; kt < Kt; ++kt) {
    v16bf af = frag_a_concat(A1, K1, A2, K2, kt*32, lane);
    acc = wmma_bf16(af, frag_packed(Bp, Kt, wid, kt, lane), acc);
  }
  int half = lane >> 4, n = wid*16 + (lane & 15);
  float bv = bias[n];
#pragma unroll
  for (int r = 0; r < 8; ++r) {
    float v = acc[r] + bv;
    if (act) v = tanhf(v);
    C[(size_t)(half*8 + r)*N + n] = v;
  }
}

// ---------- decoder gate GEMM: gates(16x2048) = [emb[tgt],O]@Wdec_ih^T + h@Wdec_hh^T + biases ----------
__global__ __launch_bounds__(256) void k_dec_gates(
  const int* formulas, const float* emb, const float* O, const float* hdec,
  const bf16_t* WdihP, const bf16_t* WdhhP,
  const float* b_ih, const float* b_hh, float* decg, int t)
{
  __shared__ float s_inp[16*352];
  int tid = threadIdx.x;
  for (int i = tid; i < 16*352; i += 256) {
    int m = i / 352, k = i % 352;
    float v = 0.0f;
    if (k < EE) {
      int tgt = formulas[m*TT + t];
      v = emb[(size_t)tgt*EE + k];
    } else if (k < EE + EHH) {
      v = O[m*EHH + (k - EE)];
    }
    s_inp[i] = v;
  }
  __syncthreads();
  int lane = tid & 31, wave = tid >> 5;
  int nt = blockIdx.x*8 + wave;           // 128 N-tiles of 2048
  v8f acc = {};
  for (int kt = 0; kt < 11; ++kt) {       // Kpad = 352
    v16bf af = frag_a_f32(s_inp, 352, 0, kt*32, lane);
    acc = wmma_bf16(af, frag_packed(WdihP, 11, nt, kt, lane), acc);
  }
  for (int kt = 0; kt < 16; ++kt) {       // K = 512
    v16bf af = frag_a_f32(hdec, DHH, 0, kt*32, lane);
    acc = wmma_bf16(af, frag_packed(WdhhP, 16, nt, kt, lane), acc);
  }
  int half = lane >> 4, n = nt*16 + (lane & 15);
  float bv = b_ih[n] + b_hh[n];
#pragma unroll
  for (int r = 0; r < 8; ++r)
    decg[(size_t)(half*8 + r)*2048 + n] = acc[r] + bv;
}

// ---------- decoder LSTM cell (in-place h,c update) ----------
__global__ __launch_bounds__(256) void k_dec_cell(const float* decg, float* hdec, float* cdec) {
  int idx = blockIdx.x*256 + threadIdx.x;
  if (idx >= BB*DHH) return;
  int m = idx >> 9, j = idx & 511;
  const float* g = decg + (size_t)m*2048;
  float iv = g[j], fv = g[512 + j], gv = g[1024 + j], ov = g[1536 + j];
  float cp = cdec[idx];
  float cn = sigf(fv)*cp + sigf(iv)*tanhf(gv);
  cdec[idx] = cn;
  hdec[idx] = sigf(ov)*tanhf(cn);
}

// ---------- attention scores: e[row] = sum_d beta[d]*tanh(lin_h[b,d] + Wv_enc[row,d]) ----------
__global__ __launch_bounds__(256) void k_attn_scores(
  const float* linh, const float* wvenc, const float* beta, float* ebuf)
{
  int lane = threadIdx.x & 31, wave = threadIdx.x >> 5;
  int row = blockIdx.x*8 + wave;          // 16384 rows
  int b = row >> 10;
  const float* lh = linh + (size_t)b*DHH;
  const float* wv = wvenc + (size_t)row*DHH;
  float s = 0.0f;
  for (int d = lane; d < DHH; d += 32)
    s += beta[d] * tanhf(lh[d] + wv[d]);
#pragma unroll
  for (int off = 16; off > 0; off >>= 1) s += __shfl_down(s, off, 32);
  if (lane == 0) ebuf[row] = s;
}

// ---------- softmax over 1024 positions + context = alpha @ enc_out ----------
__global__ __launch_bounds__(256) void k_softmax_ctx(const float* ebuf, const float* encf, float* ctx) {
  int b = blockIdx.x, tid = threadIdx.x;
  __shared__ float sa[PP];
  __shared__ float red[256];
  float lmax = -INFINITY;
  for (int p = tid; p < PP; p += 256) { float v = ebuf[b*PP + p]; sa[p] = v; lmax = fmaxf(lmax, v); }
  red[tid] = lmax; __syncthreads();
  for (int s = 128; s > 0; s >>= 1) { if (tid < s) red[tid] = fmaxf(red[tid], red[tid+s]); __syncthreads(); }
  float mx = red[0]; __syncthreads();
  float lsum = 0.0f;
  for (int p = tid; p < PP; p += 256) { float v = expf(sa[p] - mx); sa[p] = v; lsum += v; }
  red[tid] = lsum; __syncthreads();
  for (int s = 128; s > 0; s >>= 1) { if (tid < s) red[tid] += red[tid+s]; __syncthreads(); }
  float inv = 1.0f / red[0]; __syncthreads();
  for (int p = tid; p < PP; p += 256) sa[p] *= inv;
  __syncthreads();
  for (int d = tid; d < 512; d += 256) {
    float acc = 0.0f;
    const float* ef = encf + (size_t)b*PP*512 + d;
    for (int p = 0; p < PP; ++p) acc += sa[p] * ef[(size_t)p*512];
    ctx[b*512 + d] = acc;
  }
}

// ---------- output layer: softmax(O @ Wout^T + b) + argmax ----------
__global__ __launch_bounds__(256) void k_logits(
  const float* O, const float* Woutw, const float* Woutb, float* out, int t)
{
  int b = blockIdx.x, tid = threadIdx.x;
  __shared__ float z[VV];
  __shared__ float red[256];
  __shared__ int  redi[256];
  float lmax = -INFINITY; int lidx = 0;
  const float* orow = O + b*EHH;
  for (int v = tid; v < VV; v += 256) {
    float dot = Woutb[v];
    const float* wr = Woutw + (size_t)v*EHH;
    for (int k = 0; k < EHH; ++k) dot += orow[k]*wr[k];
    z[v] = dot;
    if (dot > lmax) { lmax = dot; lidx = v; }
  }
  red[tid] = lmax; redi[tid] = lidx; __syncthreads();
  for (int s = 128; s > 0; s >>= 1) {
    if (tid < s) {
      if (red[tid+s] > red[tid] || (red[tid+s] == red[tid] && redi[tid+s] < redi[tid])) {
        red[tid] = red[tid+s]; redi[tid] = redi[tid+s];
      }
    }
    __syncthreads();
  }
  float mx = red[0]; int amax = redi[0]; __syncthreads();
  float lsum = 0.0f;
  for (int v = tid; v < VV; v += 256) lsum += expf(z[v] - mx);
  red[tid] = lsum; __syncthreads();
  for (int s = 128; s > 0; s >>= 1) { if (tid < s) red[tid] += red[tid+s]; __syncthreads(); }
  float inv = 1.0f / red[0];
  float* lo = out + ((size_t)b*TT + t)*VV;
  for (int v = tid; v < VV; v += 256) lo[v] = expf(z[v] - mx) * inv;
  if (tid == 0) out[(size_t)BB*TT*VV + b*TT + t] = (float)amax;
}

// ================= host orchestration =================
extern "C" void kernel_launch(void* const* d_in, const int* in_sizes, int n_in,
                              void* d_out, int out_size, void* d_ws, size_t ws_size,
                              hipStream_t stream) {
  (void)in_sizes; (void)n_in; (void)out_size; (void)ws_size;
  const float* features = (const float*)d_in[0];
  const int*   formulas = (const int*)d_in[1];
  const float* Wih_f = (const float*)d_in[2];
  const float* Whh_f = (const float*)d_in[3];
  const float* bih_f = (const float*)d_in[4];
  const float* bhh_f = (const float*)d_in[5];
  const float* Wih_b = (const float*)d_in[6];
  const float* Whh_b = (const float*)d_in[7];
  const float* bih_b = (const float*)d_in[8];
  const float* bhh_b = (const float*)d_in[9];
  const float* emb   = (const float*)d_in[10];
  const float* Wc_w  = (const float*)d_in[11];
  const float* Wc_b  = (const float*)d_in[12];
  const float* Wout_w= (const float*)d_in[13];
  const float* Wout_b= (const float*)d_in[14];
  const float* Vh0   = (const float*)d_in[15];
  const float* Vc0   = (const float*)d_in[16];
  const float* beta  = (const float*)d_in[17];
  const float* Wh_w  = (const float*)d_in[18];
  const float* Wh_b  = (const float*)d_in[19];
  const float* Wv_w  = (const float*)d_in[20];
  const float* Wv_b  = (const float*)d_in[21];
  const float* Wdec_ih = (const float*)d_in[22];
  const float* Wdec_hh = (const float*)d_in[23];
  const float* bdec_ih = (const float*)d_in[24];
  const float* bdec_hh = (const float*)d_in[25];
  float* out = (float*)d_out;

  char* ws = (char*)d_ws;
  size_t off = 0;
  auto take = [&](size_t bytes) -> char* {
    char* p = ws + off;
    off = (off + bytes + 255) & ~(size_t)255;
    return p;
  };
  bf16_t* xp    = (bf16_t*)take((size_t)LL*CC*2);
  bf16_t* wihfP = (bf16_t*)take((size_t)512*1024*2);
  bf16_t* whhfP = (bf16_t*)take((size_t)256*1024*2);
  bf16_t* wihbP = (bf16_t*)take((size_t)512*1024*2);
  bf16_t* whhbP = (bf16_t*)take((size_t)256*1024*2);
  bf16_t* wvP   = (bf16_t*)take((size_t)512*512*2);
  bf16_t* whP   = (bf16_t*)take((size_t)512*512*2);
  bf16_t* wcP   = (bf16_t*)take((size_t)1024*256*2);
  bf16_t* wdihP = (bf16_t*)take((size_t)352*2048*2);
  bf16_t* wdhhP = (bf16_t*)take((size_t)512*2048*2);
  float* Gf    = (float*)take((size_t)WW*BH*1024*4);
  float* Gb    = (float*)take((size_t)WW*BH*1024*4);
  float* h_a   = (float*)take((size_t)2*BH*EHH*4);
  float* h_b   = (float*)take((size_t)2*BH*EHH*4);
  float* ccur  = (float*)take((size_t)2*BH*EHH*4);
  float* hs    = (float*)take((size_t)2*WW*BH*EHH*4);
  float* encf  = (float*)take((size_t)LL*512*4);
  bf16_t* encp = (bf16_t*)take((size_t)LL*512*2);
  float* wvenc = (float*)take((size_t)LL*512*4);
  float* hdec  = (float*)take((size_t)BB*DHH*4);
  float* cdec  = (float*)take((size_t)BB*DHH*4);
  float* Obuf  = (float*)take((size_t)BB*EHH*4);
  float* ctxb  = (float*)take((size_t)BB*512*4);
  float* linh  = (float*)take((size_t)BB*DHH*4);
  float* ebuf  = (float*)take((size_t)BB*PP*4);
  float* decg  = (float*)take((size_t)BB*2048*4);

  // ---- weight conversion into fragment-major packed layout ----
  auto cvt = [&](const float* src, bf16_t* dst, int N, int K, int Kpad) {
    int total = Kpad * N;
    k_wpack<<<(total + 255)/256, 256, 0, stream>>>(src, dst, N, K, Kpad);
  };
  cvt(Wih_f, wihfP, 1024, 512, 512);
  cvt(Whh_f, whhfP, 1024, 256, 256);
  cvt(Wih_b, wihbP, 1024, 512, 512);
  cvt(Whh_b, whhbP, 1024, 256, 256);
  cvt(Wv_w,  wvP,   512,  512, 512);
  cvt(Wh_w,  whP,   512,  512, 512);
  cvt(Wc_w,  wcP,   256, 1024, 1024);
  cvt(Wdec_ih, wdihP, 2048, 336, 352);
  cvt(Wdec_hh, wdhhP, 2048, 512, 512);

  // ---- pack input sequence into fragment-major A layout ----
  k_pack_x<<<(LL*CC + 255)/256, 256, 0, stream>>>(features, xp);

  // ---- big WMMA GEMMs: time-invariant input projections (bias folded in) ----
  // Gf/Gb: (16384 x 512) @ (512 x 1024)
  k_gemm_pack<<<(LL/16)*(1024/64)/8, 256, 0, stream>>>(xp, wihfP, bih_f, bhh_f, Gf, 1024, LL/16, 1024/64, 512/32);
  k_gemm_pack<<<(LL/16)*(1024/64)/8, 256, 0, stream>>>(xp, wihbP, bih_b, bhh_b, Gb, 1024, LL/16, 1024/64, 512/32);

  // ---- encoder bidirectional scan (double-buffered h) ----
  k_init_hc<<<(2*BH*EHH + 255)/256, 256, 0, stream>>>(Vh0, Vc0, h_a, ccur);
  for (int s = 0; s < WW; ++s) {
    const float* hp = (s & 1) ? h_b : h_a;
    float*       hn = (s & 1) ? h_a : h_b;
    k_enc_step<<<128, 256, 0, stream>>>(Gf, Gb, whhfP, whhbP, hp, hn, ccur, hs, s);
  }
  // after 64 steps final h is in h_a
  k_pack_enc<<<(LL*512 + 255)/256, 256, 0, stream>>>(hs, encf, encp);

  // Wv_enc = enc_out @ Wv^T + Wv_b : (16384 x 512) @ (512 x 512)
  k_gemm_pack<<<(LL/16)*(512/64)/8, 256, 0, stream>>>(encp, wvP, Wv_b, nullptr, wvenc, 512, LL/16, 512/64, 512/32);

  // ---- decoder init ----
  k_h0c0<<<(BB*DHH + 255)/256, 256, 0, stream>>>(h_a, ccur, hdec, cdec);
  k_ctx0<<<BB, 256, 0, stream>>>(encf, ctxb);
  // O0 = tanh([h0, ctx0] @ Wc^T + Wc_b) : K=1024, N=256 -> 16 tiles -> 2 blocks
  k_gemm_m16<<<2, 256, 0, stream>>>(hdec, DHH, ctxb, 512, wcP, 256, Wc_b, Obuf, 1);

  // ---- decoder scan (128 steps) ----
  for (int t = 0; t < TT; ++t) {
    // gates from OLD h, OLD O
    k_dec_gates<<<16, 256, 0, stream>>>(formulas, emb, Obuf, hdec, wdihP, wdhhP, bdec_ih, bdec_hh, decg, t);
    // lin_h from OLD h : (16x512)@(512x512) -> 32 tiles -> 4 blocks
    k_gemm_m16<<<4, 256, 0, stream>>>(hdec, DHH, nullptr, 0, whP, 512, Wh_b, linh, 0);
    // LSTM cell -> new h, c
    k_dec_cell<<<(BB*DHH + 255)/256, 256, 0, stream>>>(decg, hdec, cdec);
    // attention scores over 16384 rows
    k_attn_scores<<<LL/8, 256, 0, stream>>>(linh, wvenc, beta, ebuf);
    // softmax + context
    k_softmax_ctx<<<BB, 256, 0, stream>>>(ebuf, encf, ctxb);
    // O_t = tanh([h_t, context] @ Wc^T + Wc_b)
    k_gemm_m16<<<2, 256, 0, stream>>>(hdec, DHH, ctxb, 512, wcP, 256, Wc_b, Obuf, 1);
    // logits + argmax
    k_logits<<<BB, 256, 0, stream>>>(Obuf, Wout_w, Wout_b, out, t);
  }
}
